// DeformConv2d_79336635892378
// MI455X (gfx1250) — matrix operand order repaired and checked by reading, not compiled, not run on hardware
//
#include <hip/hip_runtime.h>
#include <hip/hip_bf16.h>

// ---------------------------------------------------------------------------
// Modulated deformable conv (DCNv2), fused for gfx1250 / MI455X.
// B=8, C=256, H=W=64, K=9 taps, Co=256.
//   kernel 1: repack conv weights fp32 -> bf16 A-fragments (WMMA layout)
//   kernel 2: mask branch (dw3x3 -> relu -> pw1x1 -> sigmoid)
//   kernel 3: fused bilinear-gather -> LDS bf16 B-panel -> v_wmma_f32_16x16x32_bf16
// ---------------------------------------------------------------------------

typedef __attribute__((ext_vector_type(16))) __bf16 v16bf;
typedef __attribute__((ext_vector_type(8)))  float  v8f;

#define CC   256      // input channels
#define CO   256      // output channels
#define HH   64
#define WW   64
#define HW   4096
#define NTAP 9
#define CK   (NTAP * CC)          // 2304 reduction length
#define NQ   (NTAP * (CC / 32))   // 72 K-chunks of 32
#define NMT  (CO / 16)            // 16 M-tiles

__device__ __forceinline__ unsigned int pack_bf2(float a, float b) {
  unsigned int ua = __float_as_uint(a);
  unsigned int ub = __float_as_uint(b);
  unsigned int ra = (ua + 0x7FFFu + ((ua >> 16) & 1u)) >> 16;  // RNE
  unsigned int rb = (ub + 0x7FFFu + ((ub >> 16) & 1u)) >> 16;
  return (ra & 0xFFFFu) | ((rb & 0xFFFFu) << 16);
}

// ---------------------------------------------------------------------------
// Kernel 1: conv_w [Co,C,3,3] fp32 -> bf16 A-fragments.
// wsA[((mt*NQ + q)*32 + lane)*8 + v] : uint32 = {bf16(K=kl), bf16(K=kl+1)}
// A 16x32 k order used here (contiguous, same as B panel): lane = g*16+m; VGPR v: Klocal = g*16 + v*2
// CK order: q = k*8 + cb ; Kglobal = k*256 + cb*32 + Klocal   (c-within-tap)
// ---------------------------------------------------------------------------
__global__ __launch_bounds__(256) void prep_w(const float* __restrict__ conv_w,
                                              unsigned int* __restrict__ wsA) {
  int id = blockIdx.x * 256 + threadIdx.x;        // < NMT*NQ*32*8 = 294912
  int v    = id & 7;
  int lane = (id >> 3) & 31;
  int rest = id >> 8;                             // mt*NQ + q
  int q    = rest % NQ;
  int mt   = rest / NQ;
  int m  = lane & 15;
  int g  = lane >> 4;
  int kl = g * 16 + v * 2;
  int k  = q >> 3;                                // tap 0..8
  int c  = (q & 7) * 32 + kl;                     // channel
  int o  = mt * 16 + m;
  float w0 = conv_w[(o * CC + c) * NTAP + k];
  float w1 = conv_w[(o * CC + c + 1) * NTAP + k];
  wsA[id] = pack_bf2(w0, w1);
}

// ---------------------------------------------------------------------------
// Kernel 2: mask = sigmoid(pw(relu(dw(x)))), [B,9,H,W] fp32 to workspace.
// One thread per (b,h,w) pixel, loop over 256 channels.
// ---------------------------------------------------------------------------
__global__ __launch_bounds__(256) void mask_kernel(const float* __restrict__ x,
                                                   const float* __restrict__ dww,
                                                   const float* __restrict__ pww,
                                                   const float* __restrict__ pwb,
                                                   float* __restrict__ maskOut) {
  int id = blockIdx.x * 256 + threadIdx.x;        // < B*HW = 32768
  int b = id >> 12;
  int p = id & (HW - 1);
  int h = p >> 6;
  int w = p & 63;
  const float* xb = x + (size_t)b * CC * HW;
  float a[NTAP];
#pragma unroll
  for (int t = 0; t < NTAP; ++t) a[t] = 0.f;
  for (int c = 0; c < CC; ++c) {
    const float* xc = xb + c * HW;
    float s = 0.f;
#pragma unroll
    for (int j = 0; j < 9; ++j) {
      int yy = h + (j / 3) - 1;
      int xx = w + (j % 3) - 1;
      float vv = (yy >= 0 && yy < HH && xx >= 0 && xx < WW) ? xc[yy * WW + xx] : 0.f;
      s = fmaf(vv, dww[c * 9 + j], s);
    }
    s = fmaxf(s, 0.f);
#pragma unroll
    for (int t = 0; t < NTAP; ++t) a[t] = fmaf(s, pww[t * CC + c], a[t]);
  }
#pragma unroll
  for (int t = 0; t < NTAP; ++t) {
    float z = a[t] + pwb[t];
    maskOut[((size_t)b * NTAP + t) * HW + p] = 1.f / (1.f + __expf(-z));
  }
}

// ---------------------------------------------------------------------------
// Kernel 3: fused deformable gather + GEMM.
// Block = (batch b, image row h) -> 64 pixels, all 256 outputs, CK=2304.
// 8 waves; wave wv owns M-tiles {2wv, 2wv+1} x N-tiles {0..3}.
// ---------------------------------------------------------------------------
__global__ __launch_bounds__(256, 1) void dcn_gemm(const float* __restrict__ x,
                                                   const float* __restrict__ offset,
                                                   const float* __restrict__ maskW,
                                                   const unsigned int* __restrict__ wsA,
                                                   const float* __restrict__ conv_b,
                                                   float* __restrict__ out) {
  __shared__ int          sIdx[NTAP * 64][4];                 // 9216 B
  __shared__ float        sWgt[NTAP * 64][4];                 // 9216 B
  __shared__ alignas(32) unsigned int sPanel[1024];           // 4 KB B-panel (32c x 64px bf16)

  const int b   = blockIdx.x >> 6;                 // 0..7
  const int h   = blockIdx.x & 63;                 // row
  const int tid = threadIdx.x;
  const float* xb = x + (size_t)b * CC * HW;

  // -------- Phase A: per-(tap,pixel) bilinear corners, weights folded with mask+validity
  for (int i = tid; i < NTAP * 64; i += 256) {
    int kk = i >> 6;                               // tap
    int p  = i & 63;                               // column (w)
    float dy = offset[(((size_t)b * 18 + 2 * kk) * HH + h) * WW + p];
    float dx = offset[(((size_t)b * 18 + 2 * kk + 1) * HH + h) * WW + p];
    float py = (float)(kk / 3 - 1 + h) + dy;
    float px = (float)(kk % 3 - 1 + p) + dx;
    float y0f = floorf(py), x0f = floorf(px);
    float wy = py - y0f, wx = px - x0f;
    int y0 = (int)y0f, x0 = (int)x0f;
    float mval = maskW[((size_t)b * NTAP + kk) * HW + h * WW + p];
    float cw0 = (1.f - wy) * (1.f - wx) * mval;
    float cw1 = (1.f - wy) * wx * mval;
    float cw2 = wy * (1.f - wx) * mval;
    float cw3 = wy * wx * mval;
    float cw[4] = {cw0, cw1, cw2, cw3};
#pragma unroll
    for (int j = 0; j < 4; ++j) {
      int iy = y0 + (j >> 1);
      int ix = x0 + (j & 1);
      bool valid = (iy >= 0) && (iy < HH) && (ix >= 0) && (ix < WW);
      int cy = min(max(iy, 0), HH - 1);
      int cx = min(max(ix, 0), WW - 1);
      sIdx[i][j] = cy * WW + cx;
      sWgt[i][j] = valid ? cw[j] : 0.f;
    }
  }
  __syncthreads();

  const int wv = tid >> 5;
  const int lane = tid & 31;

  v8f acc[2][4];
#pragma unroll
  for (int i = 0; i < 2; ++i)
#pragma unroll
    for (int j = 0; j < 4; ++j)
#pragma unroll
      for (int r = 0; r < 8; ++r) acc[i][j][r] = 0.f;

  const v16bf* Ag = reinterpret_cast<const v16bf*>(wsA);
  const v16bf* Bp = reinterpret_cast<const v16bf*>(sPanel);

  for (int q = 0; q < NQ; ++q) {
    const int k  = q >> 3;                         // tap
    const int cb = q & 7;                          // channel block of 32
    // ---- cooperative gather: build bf16 B-panel (fragment layout) in LDS
#pragma unroll
    for (int it = 0; it < 4; ++it) {
      int u  = tid + it * 256;                     // 0..1023 = (pixel, c-pair)
      int p  = u >> 4;
      int cp = u & 15;                             // c-pair: channels 2cp, 2cp+1
      int c  = cb * 32 + cp * 2;
      int item = k * 64 + p;
      int i0 = sIdx[item][0], i1 = sIdx[item][1], i2 = sIdx[item][2], i3 = sIdx[item][3];
      float w0 = sWgt[item][0], w1 = sWgt[item][1], w2 = sWgt[item][2], w3 = sWgt[item][3];
      const float* xc0 = xb + (size_t)c * HW;
      const float* xc1 = xc0 + HW;
      float v0 = w0 * xc0[i0] + w1 * xc0[i1] + w2 * xc0[i2] + w3 * xc0[i3];
      float v1 = w0 * xc1[i0] + w1 * xc1[i1] + w2 * xc1[i2] + w3 * xc1[i3];
      // B-fragment address: lanes 0-15 hold K=0-15, lanes 16-31 hold K=16-31
      int nt = p >> 4;
      int n  = p & 15;
      int g  = (cp >= 8) ? 1 : 0;
      int lw = g * 16 + n;
      int vv = cp - 8 * g;
      sPanel[(nt * 32 + lw) * 8 + vv] = pack_bf2(v0, v1);
    }
    __syncthreads();

    // ---- A fragments (global, pre-packed) + B fragments (LDS) + 8 WMMAs
    v16bf a0 = Ag[((size_t)(wv * 2 + 0) * NQ + q) * 32 + lane];
    v16bf a1 = Ag[((size_t)(wv * 2 + 1) * NQ + q) * 32 + lane];
    v16bf b0 = Bp[0 * 32 + lane];
    v16bf b1 = Bp[1 * 32 + lane];
    v16bf b2 = Bp[2 * 32 + lane];
    v16bf b3 = Bp[3 * 32 + lane];

    acc[0][0] = __builtin_amdgcn_wmma_f32_16x16x32_bf16(false, a0, false, b0, (short)0, acc[0][0], false, false);
    acc[0][1] = __builtin_amdgcn_wmma_f32_16x16x32_bf16(false, a0, false, b1, (short)0, acc[0][1], false, false);
    acc[0][2] = __builtin_amdgcn_wmma_f32_16x16x32_bf16(false, a0, false, b2, (short)0, acc[0][2], false, false);
    acc[0][3] = __builtin_amdgcn_wmma_f32_16x16x32_bf16(false, a0, false, b3, (short)0, acc[0][3], false, false);
    acc[1][0] = __builtin_amdgcn_wmma_f32_16x16x32_bf16(false, a1, false, b0, (short)0, acc[1][0], false, false);
    acc[1][1] = __builtin_amdgcn_wmma_f32_16x16x32_bf16(false, a1, false, b1, (short)0, acc[1][1], false, false);
    acc[1][2] = __builtin_amdgcn_wmma_f32_16x16x32_bf16(false, a1, false, b2, (short)0, acc[1][2], false, false);
    acc[1][3] = __builtin_amdgcn_wmma_f32_16x16x32_bf16(false, a1, false, b3, (short)0, acc[1][3], false, false);
    __syncthreads();
  }

  // -------- epilogue: C/D layout VGPR r -> M = r + 8*(lane>=16), N = lane&15
  const int g = lane >> 4;
  const int n = lane & 15;
#pragma unroll
  for (int t2 = 0; t2 < 2; ++t2) {
#pragma unroll
    for (int nt = 0; nt < 4; ++nt) {
#pragma unroll
      for (int r = 0; r < 8; ++r) {
        int o   = (wv * 2 + t2) * 16 + g * 8 + r;
        int pix = nt * 16 + n;
        out[((size_t)b * CO + o) * HW + h * WW + pix] = acc[t2][nt][r] + conv_b[o];
      }
    }
  }
}

// ---------------------------------------------------------------------------
extern "C" void kernel_launch(void* const* d_in, const int* in_sizes, int n_in,
                              void* d_out, int out_size, void* d_ws, size_t ws_size,
                              hipStream_t stream) {
  const float* x       = (const float*)d_in[0];
  const float* offset  = (const float*)d_in[1];
  const float* dww     = (const float*)d_in[2];
  const float* pww     = (const float*)d_in[3];
  const float* pwb     = (const float*)d_in[4];
  const float* conv_w  = (const float*)d_in[5];
  const float* conv_b  = (const float*)d_in[6];
  float* out           = (float*)d_out;

  // workspace layout: [A fragments bf16][mask fp32]
  unsigned int* wsA = (unsigned int*)d_ws;
  const size_t wsA_bytes = (size_t)NMT * NQ * 32 * 8 * 4;   // 1,179,648 B
  float* maskW = (float*)((char*)d_ws + wsA_bytes);         // 8*9*4096 fp32

  prep_w<<<(NMT * NQ * 32 * 8) / 256, 256, 0, stream>>>(conv_w, wsA);
  mask_kernel<<<(8 * HW) / 256, 256, 0, stream>>>(x, dww, pww, pwb, maskW);
  dcn_gemm<<<8 * HH, 256, 0, stream>>>(x, offset, maskW, wsA, conv_b, out);
}
